// BiLinearMHSLayer_39779987096210
// MI455X (gfx1250) — matrix operand order repaired and checked by reading, not compiled, not run on hardware
//
#include <hip/hip_runtime.h>
#include <hip/hip_bf16.h>

// Problem constants
#define B_   8
#define S_   1024
#define IN_  768
#define E_   256
#define L_   12
#define F_   (E_ * L_)   // 3072
#define BS_  (B_ * S_)   // 8192
#define SL_  (S_ * L_)   // 12288

#define WAVES 4          // waves per block (128 threads, wave32)

typedef _Float16 v16h __attribute__((ext_vector_type(16)));
typedef _Float16 v8h  __attribute__((ext_vector_type(8)));
typedef float    v8f  __attribute__((ext_vector_type(8)));
typedef float    v4f  __attribute__((ext_vector_type(4)));

// ---------------------------------------------------------------------------
// WMMA 16x16x32 f16 fragment loaders (contiguous-k order, same for A and B, wave32).
//
// A (16x32, MxK): lane l -> row m = l&15, kh = l>>4.
//   halves 0..15 hold K = kbase + kh*16 + {0..15} (contiguous)
//   (caller pre-offsets pointer by row*ld + kh*16)
//
// B (32x16, KxN): lane l -> col n = l&15, kh = l>>4.
//   halves 0..15 hold K = kbase + kh*16 + {0..15} (contiguous)
//   (caller pre-offsets pointer by col*ld + kh*16)
// ---------------------------------------------------------------------------

__device__ __forceinline__ v16h fragA_f16(const _Float16* __restrict__ p) {
  v8h lo = *(const v8h*)(p);
  v8h hi = *(const v8h*)(p + 8);
  v16h r;
#pragma unroll
  for (int i = 0; i < 8; ++i) { r[i] = lo[i]; r[8 + i] = hi[i]; }
  return r;
}

__device__ __forceinline__ v16h fragB_f16(const _Float16* __restrict__ p) {
  v8h lo = *(const v8h*)(p);
  v8h hi = *(const v8h*)(p + 8);
  v16h r;
#pragma unroll
  for (int i = 0; i < 8; ++i) { r[i] = lo[i]; r[8 + i] = hi[i]; }
  return r;
}

#define WMMA_F16(a, b, c) \
  __builtin_amdgcn_wmma_f32_16x16x32_f16(false, (a), false, (b), (short)0, (c), false, false)

// ---------------------------------------------------------------------------
// Prep: one-pass f32 -> f16 conversion (input, fc_w, bi_w). Keeps all cvt
// VALU out of the GEMM hot loops and halves operand bytes (L2-resident).
// ---------------------------------------------------------------------------
__global__ void __launch_bounds__(256)
cvt_f32_f16_kernel(const float* __restrict__ src, _Float16* __restrict__ dst, int n8) {
  int i = blockIdx.x * blockDim.x + threadIdx.x;
  if (i >= n8) return;
  v4f a = *(const v4f*)(src + (size_t)i * 8);
  v4f b = *(const v4f*)(src + (size_t)i * 8 + 4);
  v8h o;
#pragma unroll
  for (int k = 0; k < 4; ++k) { o[k] = (_Float16)a[k]; o[4 + k] = (_Float16)b[k]; }
  *(v8h*)(dst + (size_t)i * 8) = o;
}

// ---------------------------------------------------------------------------
// Generic f16 GEMM, 16x64 tile per wave (1 A-frag x 4 B-frags, 4 accums):
//   C[16,64](f16) = A[16,K] @ Bmat[64 rows, K]^T + bias[(col)&biasMask]
// Used for stage 1 (t = input@fc_w^T + fc_b, K=768) and
//          stage 2 (bl = t@bi_w^T + bias,    K=256).
// Output goes through padded LDS -> coalesced b128 f16 stores.
// ---------------------------------------------------------------------------
__global__ void __launch_bounds__(WAVES * 32)
gemm_f16_16x64_kernel(const _Float16* __restrict__ A, int lda,
                      const _Float16* __restrict__ Bm, int ldb,
                      const float* __restrict__ bias, int biasMask,
                      _Float16* __restrict__ C, int ldc,
                      int K, int colTiles) {
  __shared__ _Float16 lds16[WAVES][16][72];   // 72 halves = 144B rows (16B mult)

  const int lane = threadIdx.x & 31;
  const int wave = threadIdx.x >> 5;
  const int tile = blockIdx.x * WAVES + wave;
  const int rowTile = tile / colTiles;
  const int colTile = tile % colTiles;
  const int row0 = rowTile * 16, col0 = colTile * 64;
  const int m  = lane & 15;
  const int kh = lane >> 4;

  v8f acc[4];
#pragma unroll
  for (int j = 0; j < 4; ++j) {
    const float bv = bias[(col0 + j * 16 + m) & biasMask];
#pragma unroll
    for (int v = 0; v < 8; ++v) acc[j][v] = bv;
  }

  const _Float16* abase = A  + (size_t)(row0 + m) * lda + kh * 16;
  const _Float16* bbase = Bm + (size_t)(col0 + m) * ldb + kh * 16;

#pragma unroll 4
  for (int kb = 0; kb < K; kb += 32) {
    v16h a = fragA_f16(abase + kb);
#pragma unroll
    for (int j = 0; j < 4; ++j) {
      v16h bf = fragB_f16(bbase + (size_t)j * 16 * ldb + kb);
      acc[j] = WMMA_F16(a, bf, acc[j]);
    }
  }

  // D fragments (column-striped per lane) -> LDS
#pragma unroll
  for (int j = 0; j < 4; ++j)
#pragma unroll
    for (int v = 0; v < 8; ++v)
      lds16[wave][v + 8 * kh][j * 16 + m] = (_Float16)acc[j][v];
  __syncthreads();

  // Coalesced stores: 4 passes x (4 rows x 8 chunks of 8 halves)
#pragma unroll
  for (int pass = 0; pass < 4; ++pass) {
    const int r  = pass * 4 + (lane >> 3);
    const int ch = (lane & 7) * 8;
    v8h s = *(const v8h*)&lds16[wave][r][ch];
    *(v8h*)(C + (size_t)(row0 + r) * ldc + col0 + ch) = s;
  }
}

// ---------------------------------------------------------------------------
// Stage 3 (dominant, 51.5 GFLOP): per batch b,
//   out_b[12288,1024] (f32) = BL_b[12288,256] @ T_b[1024,256]^T
// 32x64 tile per wave: 2 A-frags x 4 B-frags -> 8 accums, 64 WMMAs over K=256.
// ~21 FLOP per L2 byte of fragment traffic; 402 MB output leaves as
// coalesced float4 row bursts via a padded LDS transpose.
// ---------------------------------------------------------------------------
__global__ void __launch_bounds__(WAVES * 32)
score_gemm_kernel(const _Float16* __restrict__ bl16,
                  const _Float16* __restrict__ t16,
                  float* __restrict__ out) {
  __shared__ float lds[WAVES][32][68];   // 68 floats = 272B rows (16B mult)

  const int lane = threadIdx.x & 31;
  const int wave = threadIdx.x >> 5;
  const int b    = blockIdx.y;
  const int tile = blockIdx.x * WAVES + wave;
  const int colTiles = S_ / 64;          // 16
  const int rowTile = tile / colTiles;   // 384 row tiles of 32
  const int colTile = tile % colTiles;
  const int row0 = rowTile * 32, col0 = colTile * 64;
  const int m  = lane & 15;
  const int kh = lane >> 4;

  const _Float16* abase = bl16 + (size_t)b * S_ * F_ + (size_t)(row0 + m) * E_ + kh * 16;
  const _Float16* bbase = t16  + (size_t)b * S_ * E_ + (size_t)(col0 + m) * E_ + kh * 16;

  v8f acc[2][4] = {};
#pragma unroll
  for (int kb = 0; kb < E_; kb += 32) {
    v16h a0 = fragA_f16(abase + kb);
    v16h a1 = fragA_f16(abase + (size_t)16 * E_ + kb);
#pragma unroll
    for (int j = 0; j < 4; ++j) {
      v16h bf = fragB_f16(bbase + (size_t)j * 16 * E_ + kb);
      acc[0][j] = WMMA_F16(a0, bf, acc[0][j]);
      acc[1][j] = WMMA_F16(a1, bf, acc[1][j]);
    }
  }

  // D fragments -> LDS (lane holds column j*16+m, rows i*16 + v + 8*kh)
#pragma unroll
  for (int i = 0; i < 2; ++i)
#pragma unroll
    for (int j = 0; j < 4; ++j)
#pragma unroll
      for (int v = 0; v < 8; ++v)
        lds[wave][i * 16 + v + 8 * kh][j * 16 + m] = acc[i][j][v];
  __syncthreads();

  // Coalesced f32 stores: 8 passes x (4 rows x 8 chunks of 8 floats)
  const size_t obase = (size_t)b * SL_ * S_ + (size_t)row0 * S_ + col0;
#pragma unroll
  for (int pass = 0; pass < 8; ++pass) {
    const int r  = pass * 4 + (lane >> 3);
    const int ch = (lane & 7) * 8;
    v4f s0 = *(const v4f*)&lds[wave][r][ch];
    v4f s1 = *(const v4f*)&lds[wave][r][ch + 4];
    float* orow = out + obase + (size_t)r * S_ + ch;
    *(v4f*)(orow)     = s0;
    *(v4f*)(orow + 4) = s1;
  }
}

// ---------------------------------------------------------------------------
extern "C" void kernel_launch(void* const* d_in, const int* in_sizes, int n_in,
                              void* d_out, int out_size, void* d_ws, size_t ws_size,
                              hipStream_t stream) {
  const float* input = (const float*)d_in[0];   // [B,S,IN]
  const float* fc_w  = (const float*)d_in[1];   // [E,IN]
  const float* fc_b  = (const float*)d_in[2];   // [E]
  const float* bi_w  = (const float*)d_in[3];   // [E*L,E]
  const float* bias  = (const float*)d_in[4];   // [E]
  float*       out   = (float*)d_out;           // [B,S,L,S]

  // Workspace layout (all f16, L2-resident):
  char* ws = (char*)d_ws;
  _Float16* in16 = (_Float16*)(ws);                                   // 12.58 MB
  _Float16* w16  = (_Float16*)(ws + (size_t)BS_ * IN_ * 2);           //  0.39 MB
  _Float16* bw16 = (_Float16*)(ws + ((size_t)BS_ * IN_ + (size_t)E_ * IN_) * 2);
  _Float16* t16  = (_Float16*)(ws + ((size_t)BS_ * IN_ + (size_t)E_ * IN_ + (size_t)F_ * E_) * 2);
  _Float16* bl16 = (_Float16*)(ws + ((size_t)BS_ * IN_ + (size_t)E_ * IN_ + (size_t)F_ * E_ + (size_t)BS_ * E_) * 2);

  // Prep: f32 -> f16 one-pass conversions
  {
    int n8 = (BS_ * IN_) / 8;
    cvt_f32_f16_kernel<<<(n8 + 255) / 256, 256, 0, stream>>>(input, in16, n8);
    n8 = (E_ * IN_) / 8;
    cvt_f32_f16_kernel<<<(n8 + 255) / 256, 256, 0, stream>>>(fc_w, w16, n8);
    n8 = (F_ * E_) / 8;
    cvt_f32_f16_kernel<<<(n8 + 255) / 256, 256, 0, stream>>>(bi_w, bw16, n8);
  }

  // Stage 1: t[8192,256] = in16 @ w16^T + fc_b   (K=768, 512x4 tiles)
  {
    const int colTiles = E_ / 64;                 // 4
    const int tiles = (BS_ / 16) * colTiles;      // 2048
    gemm_f16_16x64_kernel<<<tiles / WAVES, WAVES * 32, 0, stream>>>(
        in16, IN_, w16, IN_, fc_b, E_ - 1, t16, E_, IN_, colTiles);
  }
  // Stage 2: bl[8192,3072] = t16 @ bw16^T + bias (K=256, 512x48 tiles)
  {
    const int colTiles = F_ / 64;                 // 48
    const int tiles = (BS_ / 16) * colTiles;      // 24576
    gemm_f16_16x64_kernel<<<tiles / WAVES, WAVES * 32, 0, stream>>>(
        t16, E_, bw16, E_, bias, E_ - 1, bl16, F_, E_, colTiles);
  }
  // Stage 3: per batch 384x16 = 6144 wave-tiles, grid.y = batch
  {
    const int tiles = (SL_ / 32) * (S_ / 64);     // 6144
    dim3 grid(tiles / WAVES, B_);
    score_gemm_kernel<<<grid, WAVES * 32, 0, stream>>>(bl16, t16, out);
  }
}
